// MultiHeadSelfAttention_71296457113830
// MI455X (gfx1250) — matrix operand order repaired and checked by reading, not compiled, not run on hardware
//
#include <hip/hip_runtime.h>

typedef __attribute__((ext_vector_type(16))) _Float16 v16h;
typedef __attribute__((ext_vector_type(8)))  _Float16 v8h;
typedef __attribute__((ext_vector_type(8)))  float    v8f;
typedef unsigned int u32x4 __attribute__((ext_vector_type(4)));
typedef int          i32x4 __attribute__((ext_vector_type(4)));
typedef int          i32x8 __attribute__((ext_vector_type(8)));

#define DEV __device__ __forceinline__

#if __has_builtin(__builtin_amdgcn_tensor_load_to_lds) && \
    __has_builtin(__builtin_amdgcn_s_wait_tensorcnt)
#define USE_TDM 1
#else
#define USE_TDM 0
#endif

// The TDM writes LDS through an integer address inside the D# descriptor, so
// no shader store to the __shared__ object exists and GlobalOpt folds all its
// loads to undef ("internal global with no stores"). Escaping the object's
// address INTO the asm (operand, not just a memory clobber) forces the
// compiler to assume the DMA may write it, and orders loads after the wait.
#define LDS_FILLED_BY_TDM(p) asm volatile("" :: "v"((const void*)(p)) : "memory")

// ---- WMMA fragment loaders (CDNA5 ISA 7.12.2, wave32) ----------------------
// A fragment (16x32 f16): lane = g*16 + mrow holds row `row`;
//   e[0..7] = A[row][k0+g*8 .. +7], e[8..15] = A[row][k0+16+g*8 .. +7]
DEV v16h frag_a(const _Float16* base, int ld, int row, int k0, int g) {
    const _Float16* p = base + (size_t)row * ld + k0 + g * 8;
    v8h lo = *(const v8h*)(p);
    v8h hi = *(const v8h*)(p + 16);
    v16h r;
#pragma unroll
    for (int i = 0; i < 8; ++i) { r[i] = lo[i]; r[i + 8] = hi[i]; }
    return r;
}

// B fragment (32x16 f16) from column-major storage Bt[col][k]:
//   lane holds col, e[0..7] = Bt[col][k0+g*8 .. +7], e[8..15] = Bt[col][k0+16+g*8 .. +7]
DEV v16h frag_b(const _Float16* baseT, int ld, int col, int k0, int g) {
    const _Float16* p = baseT + (size_t)col * ld + k0 + g * 8;
    v8h lo = *(const v8h*)(p);
    v8h hi = *(const v8h*)(p + 16);
    v16h r;
#pragma unroll
    for (int i = 0; i < 8; ++i) { r[i] = lo[i]; r[i + 8] = hi[i]; }
    return r;
}

DEV v8f wmma_f16(v16h a, v16h b, v8f c) {
    return __builtin_amdgcn_wmma_f32_16x16x32_f16(false, a, false, b, (short)0, c,
                                                  false, false);
}

#if USE_TDM
// ---- Tensor Data Mover: 2D f16 tile, global (row-major) -> LDS with HW pad --
// D# per CDNA5 ISA. pad codes: interval 2^(c+1)
// DWORDs between pads, amount (c+1) DWORDs inserted.
DEV void tdm_load_2d(unsigned lds_byte_off, const _Float16* gptr,
                     int rows, int rowlen, long long row_stride_elems,
                     int pad_int_code, int pad_amt_code)
{
    unsigned long long ga = (unsigned long long)gptr;
    u32x4 g0;
    g0[0] = 1u;                                                  // count=1
    g0[1] = lds_byte_off;                                        // lds_addr
    g0[2] = (unsigned)(ga & 0xffffffffull);                      // global[31:0]
    g0[3] = (unsigned)((ga >> 32) & 0x1ffffffull) | (2u << 30);  // global[56:32], type=2
    unsigned td0 = (unsigned)row_stride_elems;   // >= tile_dim0, no OOB in our use
    unsigned td1 = 1u << 30;
    unsigned long long s0 = (unsigned long long)row_stride_elems;
    i32x8 g1;
    g1[0] = (int)((1u << 16) |                    // data_size = 2 bytes
                  (1u << 20) |                    // pad_enable
                  ((unsigned)pad_int_code << 22) | ((unsigned)pad_amt_code << 25));
    g1[1] = (int)((td0 & 0xffffu) << 16);                              // dim0 lo
    g1[2] = (int)(((td0 >> 16) & 0xffffu) | ((td1 & 0xffffu) << 16));  // dim0 hi|dim1 lo
    g1[3] = (int)(((td1 >> 16) & 0xffffu) | ((unsigned)rowlen << 16)); // dim1 hi|tile0
    g1[4] = (int)((unsigned)rows & 0xffffu);                           // tile1, tile2=0
    g1[5] = (int)(s0 & 0xffffffffull);                                 // dim0_stride lo
    g1[6] = (int)((s0 >> 32) & 0xffffu);                               // stride hi
    g1[7] = 0;
    i32x4 z = {0, 0, 0, 0};
#if defined(__clang_major__) && (__clang_major__ >= 23)
    i32x8 z8 = {0, 0, 0, 0, 0, 0, 0, 0};
    __builtin_amdgcn_tensor_load_to_lds(g0, g1, z, z, z8, 0);
#else
    __builtin_amdgcn_tensor_load_to_lds(g0, g1, z, z, 0);
#endif
}
#endif  // USE_TDM

// ---- f32 -> f16 elementwise convert ----------------------------------------
__global__ __launch_bounds__(256) void convert_f16(
    const float* __restrict__ in, _Float16* __restrict__ out)
{
    size_t i = (size_t)blockIdx.x * 256 + threadIdx.x;
    out[i] = (_Float16)in[i];
}

// ---- f32 [K][N] -> f16 [N][K] tiled transpose ------------------------------
__global__ __launch_bounds__(256) void transpose_f16(
    const float* __restrict__ W, _Float16* __restrict__ Wt, int K, int N)
{
    __shared__ float tile[32][33];
    const int n0 = blockIdx.x * 32, k0 = blockIdx.y * 32;
    const int tx = threadIdx.x & 31, ty = threadIdx.x >> 5;
#pragma unroll
    for (int j = 0; j < 32; j += 8)
        tile[ty + j][tx] = W[(size_t)(k0 + ty + j) * N + n0 + tx];
    __syncthreads();
#pragma unroll
    for (int j = 0; j < 32; j += 8)
        Wt[(size_t)(n0 + ty + j) * K + k0 + tx] = (_Float16)tile[tx][ty + j];
}

// ---- GEMM: C[M,N] f32 = A[M,K] f16 @ Bt[N,K] f16 (B pre-transposed) --------
// 128x128 block tile, TK=32, TDM double-buffered LDS, 8 waves of 32x64.
__global__ __launch_bounds__(256) void gemm_f16wmma(
    const _Float16* __restrict__ A, const _Float16* __restrict__ Bt,
    float* __restrict__ C, int M, int N, int K)
{
    constexpr int TM = 128, TN = 128, TK = 32, LDK = TK + 8;   // LDS stride 40
    constexpr int AS_H = TM * LDK;                             // 5120 halfs
    constexpr int BS_H = TN * LDK;
    __shared__ __align__(16) _Float16 smem[2 * (AS_H + BS_H)]; // 40 KiB, LDS off 0

    const int tid  = threadIdx.x;
    const int lane = tid & 31;
    const int g    = lane >> 4;
    const int mrow = lane & 15;
    const int wave = tid >> 5;
    const int wm   = (wave >> 1) * 32;
    const int wn   = (wave & 1) * 64;
    const int m0   = blockIdx.y * TM;
    const int n0   = blockIdx.x * TN;
    const int nkt  = K / TK;

    v8f zero = {};
    v8f acc[2][4];
#pragma unroll
    for (int i = 0; i < 2; ++i)
#pragma unroll
        for (int j = 0; j < 4; ++j) acc[i][j] = zero;

#if USE_TDM
    if (wave == 0) {  // prologue: fill buffer 0
        tdm_load_2d(0u, A + (size_t)m0 * K, TM, TK, K, 3, 3);
        tdm_load_2d((unsigned)(AS_H * 2), Bt + (size_t)n0 * K, TN, TK, K, 3, 3);
    }
#endif

    for (int t = 0; t < nkt; ++t) {
        _Float16* As = smem + (size_t)(t & 1) * (AS_H + BS_H);
        _Float16* Bs = As + AS_H;
#if USE_TDM
        if (wave == 0) __builtin_amdgcn_s_wait_tensorcnt(0);
        __syncthreads();                        // buf(t) ready; buf(t+1) free
        LDS_FILLED_BY_TDM(smem);                // escape: DMA may write smem
        if ((t + 1 < nkt) && wave == 0) {
            unsigned nb = (unsigned)(((t + 1) & 1) * (AS_H + BS_H) * 2);
            tdm_load_2d(nb, A + (size_t)m0 * K + (size_t)(t + 1) * TK,
                        TM, TK, K, 3, 3);
            tdm_load_2d(nb + AS_H * 2, Bt + (size_t)n0 * K + (size_t)(t + 1) * TK,
                        TN, TK, K, 3, 3);
        }
#else
        __syncthreads();
        for (int i = tid; i < TM * TK; i += 256) {
            int r = i / TK, c = i % TK;
            As[r * LDK + c] = A[(size_t)(m0 + r) * K + (size_t)t * TK + c];
        }
        for (int i = tid; i < TN * TK; i += 256) {
            int r = i / TK, c = i % TK;
            Bs[r * LDK + c] = Bt[(size_t)(n0 + r) * K + (size_t)t * TK + c];
        }
        __syncthreads();
#endif
        v16h a0 = frag_a(As, LDK, wm + mrow,      0, g);
        v16h a1 = frag_a(As, LDK, wm + 16 + mrow, 0, g);
#pragma unroll
        for (int ni = 0; ni < 4; ++ni) {
            v16h b = frag_b(Bs, LDK, wn + ni * 16 + mrow, 0, g);
            acc[0][ni] = wmma_f16(a0, b, acc[0][ni]);
            acc[1][ni] = wmma_f16(a1, b, acc[1][ni]);
        }
    }
    // C/D layout: VGPR v -> M = v + g*8, N = mrow
#pragma unroll
    for (int mi = 0; mi < 2; ++mi)
#pragma unroll
        for (int ni = 0; ni < 4; ++ni)
#pragma unroll
            for (int v = 0; v < 8; ++v) {
                int row = m0 + wm + mi * 16 + v + g * 8;
                int col = n0 + wn + ni * 16 + mrow;
                C[(size_t)row * N + col] = acc[mi][ni][v];
            }
}

// ---- RoPE + split heads: qkv f32 -> q,k f16 [B,H,T,D], v f16 [B,H,D,T] -----
__global__ __launch_bounds__(256) void rope_pack(
    const float* __restrict__ qkv, const float* __restrict__ cosv,
    const float* __restrict__ sinv, _Float16* __restrict__ qh,
    _Float16* __restrict__ kh, _Float16* __restrict__ vt)
{
    size_t idx = (size_t)blockIdx.x * 256 + threadIdx.x;
    int d = (int)(idx & 127);
    int h = (int)((idx >> 7) & 15);
    int t = (int)((idx >> 11) & 2047);
    int b = (int)(idx >> 22);
    const float* base = qkv + ((size_t)b * 2048 + t) * 6144;
    int cd = h * 128 + d;
    float qv = base[cd], kv = base[2048 + cd], vv = base[4096 + cd];
    int   dro = (d < 64) ? (d + 64) : (d - 64);
    float sgn = (d < 64) ? -1.0f : 1.0f;
    float qr = base[h * 128 + dro] * sgn;
    float kr = base[2048 + h * 128 + dro] * sgn;
    float c = cosv[t * 128 + d], s = sinv[t * 128 + d];
    size_t o = (((size_t)b * 16 + h) * 2048 + t) * 128 + d;
    qh[o] = (_Float16)(qv * c + qr * s);
    kh[o] = (_Float16)(kv * c + kr * s);
    vt[(((size_t)b * 16 + h) * 128 + d) * 2048 + t] = (_Float16)vv;  // V^T
}

// ---- causal flash attention, WMMA f16, fp32 online softmax -----------------
// grid = (T/128, H, B); 8 waves, each owns 16 q rows; key blocks of 64.
__global__ __launch_bounds__(256) void flash_attn_wmma(
    const _Float16* __restrict__ qh, const _Float16* __restrict__ kh,
    const _Float16* __restrict__ vt, _Float16* __restrict__ out)
{
    constexpr int T = 2048, D = 128, H = 16, KB = 64;
    constexpr int LDD = D + 8;     // 136: Ks row stride
    constexpr int LDB = KB + 8;    // 72:  Vt / Pl row stride
    constexpr int KS_H = KB * LDD;     // 8704 halfs
    constexpr int VT_H = D * LDB;      // 9216 halfs
    __shared__ __align__(16) _Float16 smem[KS_H + VT_H + 128 * LDB];  // ~54 KiB
    _Float16* Ks = smem;               // [key][d]   (== B^T for Q.K^T)
    _Float16* Vt = smem + KS_H;        // [d][key]   (== B^T for P.V)
    _Float16* Pl = smem + KS_H + VT_H; // [qrow][key] staging for P

    const int tid  = threadIdx.x;
    const int lane = tid & 31;
    const int g    = lane >> 4;
    const int mrow = lane & 15;
    const int wave = tid >> 5;
    const int qb = blockIdx.x, h = blockIdx.y, b = blockIdx.z;
    const size_t headbase = ((size_t)(b * H + h)) * T * D;
    const int q0 = qb * 128 + wave * 16;
    const float scale = 0.08838834764831845f;  // 1/sqrt(128)

    v16h qf[4];
#pragma unroll
    for (int kc = 0; kc < 4; ++kc)
        qf[kc] = frag_a(qh + headbase, D, q0 + mrow, kc * 32, g);

    v8f zero = {};
    v8f acc[8];
    float m_run[8], l_run[8];
#pragma unroll
    for (int nd = 0; nd < 8; ++nd) acc[nd] = zero;
#pragma unroll
    for (int v = 0; v < 8; ++v) { m_run[v] = -1e30f; l_run[v] = 0.0f; }

    const int nkb = qb * 2 + 2;   // causal key-block count
    for (int kb = 0; kb < nkb; ++kb) {
        __syncthreads();          // prior reads of Ks/Vt complete
#if USE_TDM
        if (wave == 0) {
            tdm_load_2d(0u, kh + headbase + (size_t)kb * KB * D, KB, D, D, 5, 3);
            tdm_load_2d((unsigned)(KS_H * 2),
                        vt + headbase + (size_t)kb * KB, D, KB, T, 4, 3);
            __builtin_amdgcn_s_wait_tensorcnt(0);
        }
        __syncthreads();
        LDS_FILLED_BY_TDM(smem);  // escape: DMA may write smem
#else
        for (int i = tid; i < KB * (D / 8); i += 256) {
            int key = i >> 4, dc = (i & 15) * 8;
            *(v8h*)&Ks[key * LDD + dc] =
                *(const v8h*)(kh + headbase + (size_t)(kb * KB + key) * D + dc);
        }
        for (int i = tid; i < KB * D; i += 256) {
            int key = i & 63, d = i >> 6;
            Vt[d * LDB + key] = vt[headbase + (size_t)d * T + kb * KB + key];
        }
        __syncthreads();
#endif
        // S = scale * Q @ K^T (16 x 64 per wave), causal-masked
        v8f s[4];
#pragma unroll
        for (int nk = 0; nk < 4; ++nk) {
            int kg0 = kb * KB + nk * 16;
            if (kg0 > q0 + 15) {
#pragma unroll
                for (int v = 0; v < 8; ++v) s[nk][v] = -1e30f;
                continue;
            }
            v8f sv = zero;
#pragma unroll
            for (int kc = 0; kc < 4; ++kc) {
                v16h bf = frag_b(Ks, LDD, nk * 16 + mrow, kc * 32, g);
                sv = wmma_f16(qf[kc], bf, sv);
            }
            int kgl = kg0 + mrow;
#pragma unroll
            for (int v = 0; v < 8; ++v) {
                int mg = q0 + v + g * 8;
                s[nk][v] = (kgl > mg) ? -1e30f : sv[v] * scale;
            }
        }

        // Online softmax: each row lives across one 16-lane group in C-layout.
#pragma unroll
        for (int v = 0; v < 8; ++v) {
            float m = fmaxf(fmaxf(s[0][v], s[1][v]), fmaxf(s[2][v], s[3][v]));
#pragma unroll
            for (int off = 1; off < 16; off <<= 1)
                m = fmaxf(m, __shfl_xor(m, off, 32));
            float mnew  = fmaxf(m_run[v], m);
            float alpha = __expf(m_run[v] - mnew);
            m_run[v] = mnew;
            float rsum = 0.0f;
#pragma unroll
            for (int nk = 0; nk < 4; ++nk) {
                float p = __expf(s[nk][v] - mnew);
                s[nk][v] = p;
                rsum += p;
            }
#pragma unroll
            for (int off = 1; off < 16; off <<= 1)
                rsum += __shfl_xor(rsum, off, 32);
            l_run[v] = l_run[v] * alpha + rsum;
#pragma unroll
            for (int nd = 0; nd < 8; ++nd) acc[nd][v] *= alpha;
        }

        // Stage P (f16) through this wave's private LDS region -> A-fragments.
        _Float16* Pw = &Pl[(wave * 16) * LDB];
#pragma unroll
        for (int nk = 0; nk < 4; ++nk)
#pragma unroll
            for (int v = 0; v < 8; ++v)
                Pw[(v + g * 8) * LDB + nk * 16 + mrow] = (_Float16)s[nk][v];

        // O += P @ V
#pragma unroll
        for (int kc = 0; kc < 2; ++kc) {
            if (kb * KB + kc * 32 > q0 + 15) break;
            v16h pf = frag_a(Pw, LDB, mrow, kc * 32, g);
#pragma unroll
            for (int nd = 0; nd < 8; ++nd) {
                v16h bf = frag_b(Vt, LDB, nd * 16 + mrow, kc * 32, g);
                acc[nd] = wmma_f16(pf, bf, acc[nd]);
            }
        }
    }

    // Normalize; emit f16 [B,T,C] so the projection GEMM consumes it directly.
#pragma unroll
    for (int nd = 0; nd < 8; ++nd)
#pragma unroll
        for (int v = 0; v < 8; ++v) {
            int tq = q0 + v + g * 8;
            out[((size_t)b * T + tq) * 2048 + h * D + nd * 16 + mrow] =
                (_Float16)(acc[nd][v] / l_run[v]);
        }
}

// ---- host-side launch -------------------------------------------------------
extern "C" void kernel_launch(void* const* d_in, const int* in_sizes, int n_in,
                              void* d_out, int out_size, void* d_ws, size_t ws_size,
                              hipStream_t stream)
{
    (void)in_sizes; (void)n_in; (void)out_size; (void)ws_size;
    const float* x     = (const float*)d_in[0];
    const float* cosv  = (const float*)d_in[1];
    const float* sinv  = (const float*)d_in[2];
    const float* Wattn = (const float*)d_in[3];
    const float* Wproj = (const float*)d_in[4];
    float* out = (float*)d_out;

    constexpr int Bb = 2, T = 2048, C = 2048, H = 16;
    const int M = Bb * T;  // 4096

    // Workspace (~200 MB). qkv f32 is recycled as the f16 attention output.
    char* ws = (char*)d_ws;
    float*    qkv    = (float*)ws;      ws += (size_t)M * 3 * C * sizeof(float);
    _Float16* xh     = (_Float16*)ws;   ws += (size_t)M * C * sizeof(_Float16);
    _Float16* WattnT = (_Float16*)ws;   ws += (size_t)3 * C * C * sizeof(_Float16);
    _Float16* WprojT = (_Float16*)ws;   ws += (size_t)C * C * sizeof(_Float16);
    _Float16* qhp    = (_Float16*)ws;   ws += (size_t)M * C * sizeof(_Float16);
    _Float16* khp    = (_Float16*)ws;   ws += (size_t)M * C * sizeof(_Float16);
    _Float16* vtp    = (_Float16*)ws;   ws += (size_t)M * C * sizeof(_Float16);
    _Float16* attnh  = (_Float16*)qkv;  // reuse: qkv dead after rope_pack

    // 0) one-time precision/layout prep (bandwidth-trivial vs 206 GFLOP)
    convert_f16<<<(unsigned)((size_t)M * C / 256), 256, 0, stream>>>(x, xh);
    transpose_f16<<<dim3(3 * C / 32, C / 32), 256, 0, stream>>>(Wattn, WattnT, C, 3 * C);
    transpose_f16<<<dim3(C / 32, C / 32), 256, 0, stream>>>(Wproj, WprojT, C, C);

    // 1) qkv = x @ W_attn
    gemm_f16wmma<<<dim3(3 * C / 128, M / 128), 256, 0, stream>>>(
        xh, WattnT, qkv, M, 3 * C, C);

    // 2) RoPE + head split (V transposed for TDM-friendly tiles)
    rope_pack<<<(unsigned)((size_t)M * C / 256), 256, 0, stream>>>(
        qkv, cosv, sinv, qhp, khp, vtp);

    // 3) causal flash attention
    flash_attn_wmma<<<dim3(T / 128, H, Bb), 256, 0, stream>>>(qhp, khp, vtp, attnh);

    // 4) out = attn @ W_proj
    gemm_f16wmma<<<dim3(C / 128, M / 128), 256, 0, stream>>>(
        attnh, WprojT, out, M, C, C);
}
